// GCLSTM_75015898792591
// MI455X (gfx1250) — matrix operand order repaired and checked by reading, not compiled, not run on hardware
//
#include <hip/hip_runtime.h>

#define Bsz 512
#define Tsz 32
#define Hsz 300
#define HP  304            // H padded to 19*16
#define NK  10
#define TB  (Tsz*Bsz)      // 16384
#define KDIM 608           // 2*HP = 19*32
#define LDSK  624
#define LDSK6 336

typedef __bf16 v16bf __attribute__((ext_vector_type(16)));
typedef float  v8f   __attribute__((ext_vector_type(8)));
typedef unsigned short u16;
typedef unsigned int   u32;

union BFrag { v16bf v; uint4 q[2]; };
union Pack8 { u16 s[8]; uint4 q; };

__device__ __forceinline__ u16 f2bf(float f){
  u32 u = __builtin_bit_cast(u32, f);
  u += 0x7FFFu + ((u >> 16) & 1u);           // round-to-nearest-even
  return (u16)(u >> 16);
}
__device__ __forceinline__ float bf2f(u16 h){
  u32 u = ((u32)h) << 16;
  return __builtin_bit_cast(float, u);
}
__device__ __forceinline__ float sigmoidf(float x){ return 1.0f/(1.0f+expf(-x)); }

// ---------------------------------------------------------------------------
// k0: pack wp / F1 / fuse1 into bf16 A-fragment order (per-lane contiguous).
// A 16x32 bf16 layout: lane l -> m = mt*16+(l&15), khalf=l>>4,
//   element e -> k = ks*32 + 16*khalf + e.
// K-mapping: k<300 -> first half row k ; 304<=k<604 -> second half row k-304
// (stored weight row k-4) ; gaps are zero padding.
// ---------------------------------------------------------------------------
__global__ void k0_prep(const float* __restrict__ wp, const float* __restrict__ F1,
                        const float* __restrict__ fuse1,
                        u16* __restrict__ WPf, u16* __restrict__ F1f, u16* __restrict__ FUf){
  int id = blockIdx.x*256 + threadIdx.x;
  const int NWP = 19*19*512, NF1 = 13*19*512, NFU = 7*10*512;
  if (id < NWP){
    int mt = id/(19*512); int r = id%(19*512); int ks = r/512; int l=(r%512)/16; int e=r%16;
    int m = mt*16 + (l&15); int kh = l>>4;
    int k = ks*32 + 16*kh + e;
    float v = 0.f;
    if (m < Hsz){
      if (k < Hsz)                 v = wp[m*600 + k];
      else if (k >= HP && k < HP+Hsz) v = wp[m*600 + (k-4)];
    }
    WPf[id] = f2bf(v);
  } else if (id < NWP+NF1){
    int j = id - NWP;
    int mt = j/(19*512); int r = j%(19*512); int ks = r/512; int l=(r%512)/16; int e=r%16;
    int m = mt*16 + (l&15); int kh = l>>4;
    int k = ks*32 + 16*kh + e;
    float v = 0.f;
    if (m < 200){
      if (k < Hsz)                 v = F1[k*200 + m];          // F1[h][o], A[m=o][k=h]
      else if (k >= HP && k < HP+Hsz) v = F1[(k-4)*200 + m];
    }
    F1f[j] = f2bf(v);
  } else if (id < NWP+NF1+NFU){
    int j = id - NWP - NF1;
    int mt = j/(10*512); int r = j%(10*512); int ks = r/512; int l=(r%512)/16; int e=r%16;
    int m = mt*16 + (l&15); int kh = l>>4;
    int k = ks*32 + 16*kh + e;
    float v = 0.f;
    if (m < 100 && k < Hsz) v = fuse1[k*100 + m];
    FUf[j] = f2bf(v);
  }
}

// ---------------------------------------------------------------------------
// k1: LSTM cells (10 neighbors, K=11) + target cell, per (t,b) block.
// h = sigmoid(o)*tanh(sigmoid(i)*tanh(g)); forget gate unused by reference.
// Writes CELLS/HT bf16 padded to HP; t==0 copied into CON1 (con1_0 = con_0).
// ---------------------------------------------------------------------------
__global__ void k1_cells(const float* __restrict__ xg, const float* __restrict__ extras,
                         const float* __restrict__ Wih, const float* __restrict__ b_ih,
                         const float* __restrict__ b_hh, const float* __restrict__ Wt,
                         const float* __restrict__ bt_ih, const float* __restrict__ bt_hh,
                         u16* __restrict__ CELLS, u16* __restrict__ CON1, u16* __restrict__ HTb){
  int t = blockIdx.x % Tsz, b = blockIdx.x / Tsz;
  int tid = threadIdx.x;
  __shared__ float xs[110];
  __shared__ float es[4];
  const float* xb = xg + (size_t)(b*Tsz + t)*280;
  if (tid < 110) xs[tid] = xb[tid];
  if (tid >= 128 && tid < 132) es[tid-128] = extras[(size_t)(b*Tsz+t)*4 + (tid-128)];
  __syncthreads();
  size_t cb = ((size_t)(t*Bsz + b))*NK;
  for (int p = tid; p < NK*HP + HP; p += 256){
    if (p < NK*HP){
      int k = p / HP, j = p % HP;
      u16 o = 0;
      if (j < Hsz){
        const float* W = Wih + (size_t)k*1200*11;
        float gi = b_ih[k*1200 + j]       + b_hh[k*1200 + j];
        float gg = b_ih[k*1200 + 600 + j] + b_hh[k*1200 + 600 + j];
        float go = b_ih[k*1200 + 900 + j] + b_hh[k*1200 + 900 + j];
        #pragma unroll
        for (int i=0;i<11;i++){
          float xv = xs[i*10 + k];
          gi += W[(size_t)j*11 + i]*xv;
          gg += W[(size_t)(600+j)*11 + i]*xv;
          go += W[(size_t)(900+j)*11 + i]*xv;
        }
        o = f2bf(sigmoidf(go)*tanhf(sigmoidf(gi)*tanhf(gg)));
      }
      CELLS[(cb + k)*HP + j] = o;
      if (t == 0) CON1[(cb + k)*HP + j] = o;
    } else {
      int j = p - NK*HP;
      u16 o = 0;
      if (j < Hsz){
        float gi = bt_ih[j]     + bt_hh[j];
        float gg = bt_ih[600+j] + bt_hh[600+j];
        float go = bt_ih[900+j] + bt_hh[900+j];
        #pragma unroll
        for (int i=0;i<4;i++){
          float ev = es[i];
          gi += Wt[j*4+i]*ev; gg += Wt[(600+j)*4+i]*ev; go += Wt[(900+j)*4+i]*ev;
        }
        o = f2bf(sigmoidf(go)*tanhf(sigmoidf(gi)*tanhf(gg)));
      }
      HTb[(size_t)(t*Bsz+b)*HP + j] = o;
    }
  }
}

// ---------------------------------------------------------------------------
// k2: con1 = relu(wp @ [con_t ; con_{t-1}] + bp) via WMMA bf16.
// M=304 (19 tiles), K=608 (19 steps), N=31*512*10; 16 columns per block in LDS.
// ---------------------------------------------------------------------------
__global__ void k2_con1(const u16* __restrict__ WPf, const u16* __restrict__ CELLS,
                        const float* __restrict__ bp, u16* __restrict__ CON1){
  __shared__ __align__(16) u16 X[16*LDSK];
  int tid = threadIdx.x;
  int gc0 = blockIdx.x*16;                 // column = (t-1)*5120 + b*10 + k
  int tt  = gc0/5120 + 1;                  // 5120 % 16 == 0 -> one t per block
  const u32* Cu = (const u32*)CELLS;
  u32* Xu = (u32*)X;
  for (int q = tid; q < 16*HP; q += 256){  // 304 dwords per column
    int c = q / 304, wd = q % 304;
    int rem = (gc0 + c) % 5120;
    Xu[c*(LDSK/2) + wd] = (wd < 152)
      ? Cu[((size_t)tt*5120     + rem)*152 + wd]
      : Cu[((size_t)(tt-1)*5120 + rem)*152 + (wd-152)];
  }
  __syncthreads();
  int wave = tid>>5, lane = tid&31, n = lane&15, kh = lane>>4;
  v8f a0 = {0.f,0.f,0.f,0.f,0.f,0.f,0.f,0.f}, a1 = a0, a2 = a0;
  int m0 = wave, m1 = wave + 8, m2 = wave + 16;
  bool h2 = (wave < 3);                    // waves 0..2 own tiles 16..18
  for (int ks = 0; ks < 19; ks++){
    BFrag fb;
    const uint4* bq = (const uint4*)(X + n*LDSK + ks*32 + 16*kh);
    fb.q[0] = bq[0]; fb.q[1] = bq[1];
    BFrag fa;
    const uint4* aq = (const uint4*)(WPf + ((size_t)(m0*19 + ks)*32 + lane)*16);
    fa.q[0]=aq[0]; fa.q[1]=aq[1];
    a0 = __builtin_amdgcn_wmma_f32_16x16x32_bf16(false, fa.v, false, fb.v, (short)0, a0, false, false);
    aq = (const uint4*)(WPf + ((size_t)(m1*19 + ks)*32 + lane)*16);
    fa.q[0]=aq[0]; fa.q[1]=aq[1];
    a1 = __builtin_amdgcn_wmma_f32_16x16x32_bf16(false, fa.v, false, fb.v, (short)0, a1, false, false);
    if (h2){
      aq = (const uint4*)(WPf + ((size_t)(m2*19 + ks)*32 + lane)*16);
      fa.q[0]=aq[0]; fa.q[1]=aq[1];
      a2 = __builtin_amdgcn_wmma_f32_16x16x32_bf16(false, fa.v, false, fb.v, (short)0, a2, false, false);
    }
  }
  int rem = (gc0 + n) % 5120;
  size_t colbase = ((size_t)tt*5120 + rem)*HP;
  auto store_tile = [&](v8f av, int mt){
    int mb = mt*16 + 8*kh;
    Pack8 pk;
    #pragma unroll
    for (int r=0;r<8;r++){
      int M = mb + r;
      float v = av[r] + ((M < Hsz) ? bp[M] : 0.f);
      pk.s[r] = (M < Hsz) ? f2bf(fmaxf(v, 0.f)) : (u16)0;
    }
    *(uint4*)(CON1 + colbase + mb) = pk.q;   // 16B-aligned (HP*2 = 608)
  };
  store_tile(a0, m0);
  store_tile(a1, m1);
  if (h2) store_tile(a2, m2);
}

// ---------------------------------------------------------------------------
// k3: fc1 = relu(hcat@F1+b1) via WMMA, fused F2 reduction -> wdyn.
// hcat columns = [con1 (HP) ; htarget (HP)]; M=208 (13 tiles), K=608.
// ---------------------------------------------------------------------------
__global__ void k3_wdyn(const u16* __restrict__ F1f, const u16* __restrict__ CON1,
                        const u16* __restrict__ HTb, const float* __restrict__ b1,
                        const float* __restrict__ F2, const float* __restrict__ b2,
                        const float* __restrict__ xg, const float* __restrict__ AngleM,
                        float* __restrict__ WDYN){
  __shared__ __align__(16) u16 X[16*LDSK];
  __shared__ float part[8][16];
  int tid = threadIdx.x;
  int gc0 = blockIdx.x*16;                 // column = t*5120 + b*10 + k
  const u32* C1u = (const u32*)CON1;
  const u32* HTu = (const u32*)HTb;
  u32* Xu = (u32*)X;
  for (int q = tid; q < 16*HP; q += 256){
    int c = q / 304, wd = q % 304;
    int col = gc0 + c;
    Xu[c*(LDSK/2) + wd] = (wd < 152)
      ? C1u[(size_t)col*152 + wd]
      : HTu[(size_t)(col/10)*152 + (wd-152)];   // col/10 = t*512 + b
  }
  __syncthreads();
  int wave = tid>>5, lane = tid&31, n = lane&15, kh = lane>>4;
  v8f a0 = {0.f,0.f,0.f,0.f,0.f,0.f,0.f,0.f}, a1 = a0;
  int m0 = wave, m1 = wave + 8;
  bool h1 = (wave < 5);                    // 13 tiles over 8 waves
  for (int ks = 0; ks < 19; ks++){
    BFrag fb;
    const uint4* bq = (const uint4*)(X + n*LDSK + ks*32 + 16*kh);
    fb.q[0] = bq[0]; fb.q[1] = bq[1];
    BFrag fa;
    const uint4* aq = (const uint4*)(F1f + ((size_t)(m0*19 + ks)*32 + lane)*16);
    fa.q[0]=aq[0]; fa.q[1]=aq[1];
    a0 = __builtin_amdgcn_wmma_f32_16x16x32_bf16(false, fa.v, false, fb.v, (short)0, a0, false, false);
    if (h1){
      aq = (const uint4*)(F1f + ((size_t)(m1*19 + ks)*32 + lane)*16);
      fa.q[0]=aq[0]; fa.q[1]=aq[1];
      a1 = __builtin_amdgcn_wmma_f32_16x16x32_bf16(false, fa.v, false, fb.v, (short)0, a1, false, false);
    }
  }
  float s = 0.f;
  {
    int mb = m0*16 + 8*kh;
    #pragma unroll
    for (int r=0;r<8;r++){ int M = mb + r; if (M < 200) s += fmaxf(a0[r] + b1[M], 0.f)*F2[M]; }
  }
  if (h1){
    int mb = m1*16 + 8*kh;
    #pragma unroll
    for (int r=0;r<8;r++){ int M = mb + r; if (M < 200) s += fmaxf(a1[r] + b1[M], 0.f)*F2[M]; }
  }
  s += __shfl_xor(s, 16);
  if (lane < 16) part[wave][lane] = s;
  __syncthreads();
  if (tid < 16){
    float wd = 0.f;
    #pragma unroll
    for (int w2=0; w2<8; w2++) wd += part[w2][tid];
    int col = gc0 + tid;
    int t = col / 5120, rem = col % 5120, b = rem/10, k = rem%10;
    float x8  = xg[((size_t)(b*Tsz+t)*28 +  8)*10 + k];
    float x10 = xg[((size_t)(b*Tsz+t)*28 + 10)*10 + k];
    float ang = fabsf(x10 - AngleM[k]) * (1.0f/360.0f);
    WDYN[col] = fmaxf(wd + x8*F2[200] + ang*F2[201] + b2[0], 0.f);
  }
}

// ---------------------------------------------------------------------------
// k4: per-t neighbor softmax. The reference's [10,B]-softmax + flat reshape
// collapses to: WA3[t] (flat [B*10]) == S (flat [10,B]). One block per t.
// ---------------------------------------------------------------------------
__global__ void k4_softmax(const float* __restrict__ WDYN, float* __restrict__ WA3){
  int t = blockIdx.x, c = threadIdx.x;     // c = batch column, 512 threads
  __shared__ float S[5120];
  const float* wd = WDYN + (size_t)t*5120 + c*10;
  float mx = wd[0];
  for (int n2=1;n2<10;n2++) mx = fmaxf(mx, wd[n2]);
  float e[10]; float sum = 0.f;
  for (int n2=0;n2<10;n2++){ e[n2] = expf(wd[n2]-mx); sum += e[n2]; }
  float inv = 1.0f/sum;
  for (int n2=0;n2<10;n2++) S[n2*512 + c] = e[n2]*inv;
  __syncthreads();
  for (int j=0;j<10;j++){
    int f = c*10 + j;
    WA3[(size_t)t*5120 + f] = S[f];        // flat reinterpretation == the "bug"
  }
}

// ---------------------------------------------------------------------------
// k5: cat = sum_k con1*wa3 (bf16 out, K padded to 320) and the fusiondis path
// pre-dotted with Wout (sum_k wA = 1 makes the bias term exact).
// ---------------------------------------------------------------------------
__global__ void k5_cat(const float* __restrict__ xg, const float* __restrict__ DisM,
                       const float* __restrict__ ff, const float* __restrict__ bff,
                       const float* __restrict__ Wout, const u16* __restrict__ CON1,
                       const float* __restrict__ WA3, u16* __restrict__ CATb,
                       float* __restrict__ FD){
  int tb = blockIdx.x; int t = tb >> 9, b = tb & 511;
  int tid = threadIdx.x;
  __shared__ float wa3s[10], wAs[10], xws[17], red[256];
  if (tid < 10) wa3s[tid] = WA3[(size_t)t*5120 + b*10 + tid];
  if (tid == 0){
    float m = DisM[0];
    for (int k=1;k<10;k++) m = fmaxf(m, DisM[k]);
    float ev[10], s = 0.f;
    for (int k=0;k<10;k++){ ev[k] = expf(DisM[k]-m); s += ev[k]; }
    float inv = 1.f/s;
    for (int k=0;k<10;k++) wAs[k] = ev[k]*inv;
  }
  __syncthreads();
  if (tid < 17){
    float s = 0.f;
    for (int k=0;k<10;k++) s += xg[((size_t)(b*Tsz+t)*28 + 11 + tid)*10 + k]*wAs[k];
    xws[tid] = s;
  }
  __syncthreads();
  size_t cb = (size_t)tb*10;
  for (int p = tid; p < 320; p += 256){
    u16 o = 0;
    if (p < Hsz){
      float cv = 0.f;
      #pragma unroll
      for (int k=0;k<10;k++) cv += bf2f(CON1[(cb+k)*HP + p])*wa3s[k];
      o = f2bf(cv);
    }
    CATb[(size_t)tb*320 + p] = o;
  }
  float v = 0.f;
  if (tid < 100){
    float fo = bff[tid];
    #pragma unroll
    for (int f=0;f<17;f++) fo += ff[tid*17+f]*xws[f];
    v = fo*Wout[tid];
  }
  red[tid] = v;
  __syncthreads();
  for (int s2 = 128; s2 > 0; s2 >>= 1){
    if (tid < s2) red[tid] += red[tid+s2];
    __syncthreads();
  }
  if (tid == 0) FD[tb] = red[0];
}

// ---------------------------------------------------------------------------
// k6: fuse = cat@fuse1 via WMMA (M=112, K=320), fused Wout reduction and final
// mix; writes preds and labels_r.
// ---------------------------------------------------------------------------
__global__ void k6_out(const u16* __restrict__ FUf, const u16* __restrict__ CATb,
                       const float* __restrict__ biasf, const float* __restrict__ Wout,
                       const float* __restrict__ biasout, const float* __restrict__ av,
                       const float* __restrict__ FD, const float* __restrict__ labels,
                       float* __restrict__ out){
  __shared__ __align__(16) u16 X[16*LDSK6];
  __shared__ float part[8][16];
  __shared__ float bdot;
  int tid = threadIdx.x;
  int gc0 = blockIdx.x*16;                 // column = t*512 + b
  const u32* Cu = (const u32*)CATb;
  u32* Xu = (u32*)X;
  for (int q = tid; q < 16*160; q += 256){
    int c = q/160, wd = q%160;
    Xu[c*(LDSK6/2) + wd] = Cu[(size_t)(gc0+c)*160 + wd];
  }
  if (tid == 0){
    float s = 0.f;
    for (int o2=0;o2<100;o2++) s += biasf[o2]*Wout[o2];
    bdot = s;
  }
  __syncthreads();
  int wave = tid>>5, lane = tid&31, n = lane&15, kh = lane>>4;
  if (wave < 7){
    v8f acc = {0.f,0.f,0.f,0.f,0.f,0.f,0.f,0.f};
    for (int ks = 0; ks < 10; ks++){
      BFrag fb;
      const uint4* bq = (const uint4*)(X + n*LDSK6 + ks*32 + 16*kh);
      fb.q[0] = bq[0]; fb.q[1] = bq[1];
      BFrag fa;
      const uint4* aq = (const uint4*)(FUf + ((size_t)(wave*10 + ks)*32 + lane)*16);
      fa.q[0]=aq[0]; fa.q[1]=aq[1];
      acc = __builtin_amdgcn_wmma_f32_16x16x32_bf16(false, fa.v, false, fb.v, (short)0, acc, false, false);
    }
    float s = 0.f;
    int mb = wave*16 + 8*kh;
    #pragma unroll
    for (int r=0;r<8;r++){ int M = mb + r; if (M < 100) s += acc[r]*Wout[M]; }
    s += __shfl_xor(s, 16);
    if (lane < 16) part[wave][lane] = s;
  } else {
    if (lane < 16) part[7][lane] = 0.f;
  }
  __syncthreads();
  if (tid < 16){
    int col = gc0 + tid;
    float fsum = 0.f;
    #pragma unroll
    for (int w2=0; w2<8; w2++) fsum += part[w2][tid];
    float aa = av[0];
    float pred = aa*(fsum + bdot) + (1.f-aa)*FD[col] + biasout[0];
    out[col] = pred;
    int t = col >> 9, b = col & 511;
    out[TB + col] = labels[(size_t)b*Tsz + t];  // labels_r[t,b] = labels[b,t]
  }
}

// ---------------------------------------------------------------------------
extern "C" void kernel_launch(void* const* d_in, const int* in_sizes, int n_in,
                              void* d_out, int out_size, void* d_ws, size_t ws_size,
                              hipStream_t stream){
  (void)in_sizes; (void)n_in; (void)out_size; (void)ws_size;
  const float* xg      = (const float*)d_in[0];
  const float* labels  = (const float*)d_in[1];
  const float* extras  = (const float*)d_in[2];
  const float* DisM    = (const float*)d_in[3];
  const float* AngleM  = (const float*)d_in[4];
  const float* Wih     = (const float*)d_in[5];
  const float* b_ih    = (const float*)d_in[6];
  const float* b_hh    = (const float*)d_in[7];
  const float* Wt      = (const float*)d_in[8];
  const float* bt_ih   = (const float*)d_in[9];
  const float* bt_hh   = (const float*)d_in[10];
  const float* wp      = (const float*)d_in[11];
  const float* bp      = (const float*)d_in[12];
  const float* F1      = (const float*)d_in[13];
  const float* b1      = (const float*)d_in[14];
  const float* F2      = (const float*)d_in[15];
  const float* b2      = (const float*)d_in[16];
  const float* ff      = (const float*)d_in[17];
  const float* bff     = (const float*)d_in[18];
  const float* fuse1   = (const float*)d_in[19];
  const float* biasf   = (const float*)d_in[20];
  const float* Wout    = (const float*)d_in[21];
  const float* biasout = (const float*)d_in[22];
  const float* av      = (const float*)d_in[23];

  char* w = (char*)d_ws;
  size_t off = 0;
  auto take = [&](size_t bytes)->char*{
    char* p = w + off;
    off += (bytes + 255) & ~(size_t)255;
    return p;
  };
  u16*   WPf   = (u16*)  take((size_t)19*19*512*2);
  u16*   F1f   = (u16*)  take((size_t)13*19*512*2);
  u16*   FUf   = (u16*)  take((size_t)7*10*512*2);
  u16*   CELLS = (u16*)  take((size_t)TB*NK*HP*2);
  u16*   CON1  = (u16*)  take((size_t)TB*NK*HP*2);
  u16*   HTb   = (u16*)  take((size_t)TB*HP*2);
  float* WDYN  = (float*)take((size_t)TB*NK*4);
  float* WA3   = (float*)take((size_t)TB*NK*4);
  u16*   CATb  = (u16*)  take((size_t)TB*320*2);
  float* FD    = (float*)take((size_t)TB*4);

  k0_prep   <<<1356,  256, 0, stream>>>(wp, F1, fuse1, WPf, F1f, FUf);
  k1_cells  <<<TB,    256, 0, stream>>>(xg, extras, Wih, b_ih, b_hh, Wt, bt_ih, bt_hh,
                                        CELLS, CON1, HTb);
  k2_con1   <<<9920,  256, 0, stream>>>(WPf, CELLS, bp, CON1);
  k3_wdyn   <<<10240, 256, 0, stream>>>(F1f, CON1, HTb, b1, F2, b2, xg, AngleM, WDYN);
  k4_softmax<<<Tsz,   512, 0, stream>>>(WDYN, WA3);
  k5_cat    <<<TB,    256, 0, stream>>>(xg, DisM, ff, bff, Wout, CON1, WA3, CATb, FD);
  k6_out    <<<1024,  256, 0, stream>>>(FUf, CATb, biasf, Wout, biasout, av, FD, labels,
                                        (float*)d_out);
}
